// LSTMEncoderDecoderAttention_46617575031468
// MI455X (gfx1250) — matrix operand order repaired and checked by reading, not compiled, not run on hardware
//
#include <hip/hip_runtime.h>
#include <hip/hip_bf16.h>

#define H      128
#define T_IN   168
#define OUT_H  24
#define BATCH  4096

typedef __attribute__((ext_vector_type(16))) _Float16 v16h;
typedef __attribute__((ext_vector_type(8)))  float    v8f;

// ---- fast transcendental helpers (hardware TRANS ops, no libm branches) ----
__device__ __forceinline__ float fast_exp(float x) {
#if __has_builtin(__builtin_amdgcn_exp2f)
    return __builtin_amdgcn_exp2f(x * 1.44269504088896340736f);   // v_exp_f32
#else
    return __expf(x);
#endif
}
__device__ __forceinline__ float fast_rcp(float x) {
#if __has_builtin(__builtin_amdgcn_rcpf)
    return __builtin_amdgcn_rcpf(x);                              // v_rcp_f32
#else
    return 1.0f / x;
#endif
}
__device__ __forceinline__ float fast_tanh(float x) {
#if __has_builtin(__builtin_amdgcn_tanhf)
    return __builtin_amdgcn_tanhf(x);                             // v_tanh_f32
#elif __has_builtin(__builtin_amdgcn_tanh_f32)
    return __builtin_amdgcn_tanh_f32(x);
#else
    float e = fast_exp(-2.0f * x);                                // 2 trans + fma
    return (1.0f - e) * fast_rcp(1.0f + e);
#endif
}
__device__ __forceinline__ float sigmf(float x) {
    return fast_rcp(1.0f + fast_exp(-x));
}

// A-fragment K map (16-bit A 16x32, ISA 7.12.2): e<8 -> hl*8+e ; else 16+hl*8+(e-8)
// B-fragment K map (16-bit B 32x16, by ISA B layout): e<8 -> hl*8+e ; else 16+hl*8+(e-8) ; N = lane%16

// ---------------------------------------------------------------------------
// Pack a (G x K) f32 weight matrix (row g = output unit, col k = input unit)
// into WMMA B-fragment order, f16. Fragment for tile (nt,kt) of lane l:
// dst[((nt*KT+kt)*32 + l)*16 + e] = W[nt*16 + l%16][kt*32 + (e<8 ? (l/16)*8 + e : 16 + (l/16)*8 + e-8)]
// ---------------------------------------------------------------------------
__global__ void pack_b_kernel(const float* __restrict__ src, _Float16* __restrict__ dst,
                              int G, int K, int ld, int col0) {
    int idx = blockIdx.x * blockDim.x + threadIdx.x;
    int total = G * K;
    if (idx >= total) return;
    int e    = idx & 15;
    int lane = (idx >> 4) & 31;
    int r    = idx >> 9;
    int KT   = K >> 5;
    int kt   = r % KT;
    int nt   = r / KT;
    int hl   = lane >> 4;
    int n    = lane & 15;
    int k    = kt * 32 + (e < 8 ? hl * 8 + e : 16 + hl * 8 + (e - 8));
    int g    = nt * 16 + n;
    dst[idx] = (_Float16)src[(size_t)g * ld + col0 + k];
}

// ---------------------------------------------------------------------------
// Encoder: one block = 16 batch rows, 8 waves. Entire 168-step recurrence in
// one launch. h lives in LDS (f16), c lives in VGPRs. Wave jt owns hidden
// columns [jt*16, jt*16+16) and computes all 4 gate tiles for them so the
// LSTM cell update is register-local.
// ---------------------------------------------------------------------------
__global__ __launch_bounds__(256) void encoder_kernel(
    const float* __restrict__ x,
    const float* __restrict__ enc_Wih, const float* __restrict__ enc_bih,
    const float* __restrict__ enc_bhh,
    const _Float16* __restrict__ Whh_p,
    _Float16* __restrict__ enc_f16, float* __restrict__ c_ws)
{
    __shared__ float    x_s[16 * T_IN];
    __shared__ _Float16 h_s[16][H];

    const int tid  = threadIdx.x;
    const int wave = tid >> 5, lane = tid & 31;
    const int hl   = lane >> 4, n = lane & 15, m = lane & 15;
    const int jt   = wave;
    const int b0   = blockIdx.x * 16;
    const int j    = jt * 16 + n;

    for (int i = tid; i < 16 * T_IN; i += 256)
        x_s[i] = x[(size_t)(b0 + i / T_IN) * T_IN + (i % T_IN)];
    {
        _Float16* hf = &h_s[0][0];
        for (int i = tid; i < 16 * H; i += 256) hf[i] = (_Float16)0.0f;
    }
    __syncthreads();

    float wih_c[4], bias_c[4];
#pragma unroll
    for (int g = 0; g < 4; ++g) {
        int gi = g * H + j;
        wih_c[g]  = enc_Wih[gi];
        bias_c[g] = enc_bih[gi] + enc_bhh[gi];
    }
    float c_st[8];
#pragma unroll
    for (int r = 0; r < 8; ++r) c_st[r] = 0.0f;

    for (int t = 0; t < T_IN; ++t) {
        v8f acc[4];
#pragma unroll
        for (int g = 0; g < 4; ++g)
#pragma unroll
            for (int q = 0; q < 8; ++q) acc[g][q] = 0.0f;

#pragma unroll
        for (int kt = 0; kt < 4; ++kt) {
            v16h a;
            const _Float16* p1 = &h_s[m][kt * 32 + hl * 8];
            const _Float16* p2 = &h_s[m][kt * 32 + 16 + hl * 8];
#pragma unroll
            for (int e = 0; e < 8; ++e) { a[e] = p1[e]; a[8 + e] = p2[e]; }
#pragma unroll
            for (int g = 0; g < 4; ++g) {
                int nt = g * 8 + jt;
                v16h bf = *(const v16h*)(Whh_p + ((size_t)(nt * 4 + kt) * 32 + lane) * 16);
                acc[g] = __builtin_amdgcn_wmma_f32_16x16x32_f16(
                    false, a, false, bf, (short)0, acc[g], false, false);
            }
        }
        __syncthreads();   // all reads of h_s done before overwriting

#pragma unroll
        for (int r = 0; r < 8; ++r) {
            int   bl = r + 8 * hl;
            float xv = x_s[bl * T_IN + t];
            float gi_ = acc[0][r] + xv * wih_c[0] + bias_c[0];
            float gf_ = acc[1][r] + xv * wih_c[1] + bias_c[1];
            float gg_ = acc[2][r] + xv * wih_c[2] + bias_c[2];
            float go_ = acc[3][r] + xv * wih_c[3] + bias_c[3];
            float iv = sigmf(gi_), fv = sigmf(gf_);
            float gv = fast_tanh(gg_), ov = sigmf(go_);
            float cn = fv * c_st[r] + iv * gv;
            c_st[r] = cn;
            float hn = ov * fast_tanh(cn);
            h_s[bl][j] = (_Float16)hn;
            enc_f16[((size_t)(b0 + bl) * T_IN + t) * H + j] = (_Float16)hn;
        }
        __syncthreads();
    }
#pragma unroll
    for (int r = 0; r < 8; ++r)
        c_ws[(size_t)(b0 + r + 8 * hl) * H + j] = c_st[r];
}

// ---------------------------------------------------------------------------
// Ue = enc_outs @ Ua^T  (B*T x 128 @ 128 x 128), WMMA, f16 out.
// One block per 16-row tile; wave w owns output column tile w.
// ---------------------------------------------------------------------------
__global__ __launch_bounds__(256) void ue_kernel(
    const _Float16* __restrict__ enc_f16,
    const _Float16* __restrict__ Ua_p,
    _Float16* __restrict__ Ue)
{
    const int tid = threadIdx.x, wave = tid >> 5, lane = tid & 31;
    const int hl = lane >> 4, n = lane & 15, m = lane & 15;
    const size_t row0 = (size_t)blockIdx.x * 16;

    v8f acc;
#pragma unroll
    for (int q = 0; q < 8; ++q) acc[q] = 0.0f;

    const _Float16* arow = enc_f16 + (row0 + m) * H;
#pragma unroll
    for (int kt = 0; kt < 4; ++kt) {
        v16h a;
#pragma unroll
        for (int e = 0; e < 8; ++e) {
            a[e]     = arow[kt * 32 + hl * 8 + e];
            a[8 + e] = arow[kt * 32 + 16 + hl * 8 + e];
        }
        v16h bf = *(const v16h*)(Ua_p + ((size_t)(wave * 4 + kt) * 32 + lane) * 16);
        acc = __builtin_amdgcn_wmma_f32_16x16x32_f16(
            false, a, false, bf, (short)0, acc, false, false);
    }
#pragma unroll
    for (int r = 0; r < 8; ++r)
        Ue[(row0 + r + 8 * hl) * H + wave * 16 + n] = (_Float16)acc[r];
}

// ---------------------------------------------------------------------------
// Decoder: one block = 16 batch rows, full 24-step recurrence.
// Per step: q=h@Wa^T (WMMA) -> scores=v.tanh(Ue+q), softmax, context (VALU)
//           -> gates = h@Whh^T + ctx@Wih[:,1:]^T (WMMA, fused accumulators)
//           -> LSTM cell (c in VGPRs) -> pred = h2@fc^T (wave reduce).
// ---------------------------------------------------------------------------
__global__ __launch_bounds__(256) void decoder_kernel(
    const float* __restrict__ x,
    const float* __restrict__ va, const float* __restrict__ fc_W,
    const float* __restrict__ fc_b,
    const float* __restrict__ dec_Wih,
    const float* __restrict__ dec_bih, const float* __restrict__ dec_bhh,
    const _Float16* __restrict__ Wa_p, const _Float16* __restrict__ decWhh_p,
    const _Float16* __restrict__ decWih1_p,
    const _Float16* __restrict__ enc_f16, const _Float16* __restrict__ Ue,
    const float* __restrict__ c_ws,
    float* __restrict__ out)
{
    __shared__ _Float16 h_s[16][H];
    __shared__ _Float16 q_s[16][H];
    __shared__ _Float16 ctx_s[16][H];
    __shared__ float    att_s[16][T_IN];
    __shared__ float    din_s[16];
    __shared__ float    v_s[H];
    __shared__ float    fc_s[H];

    const int tid = threadIdx.x, wave = tid >> 5, lane = tid & 31;
    const int hl = lane >> 4, n = lane & 15, m = lane & 15;
    const int jt = wave;
    const int b0 = blockIdx.x * 16;
    const int j  = jt * 16 + n;

    // init: h from last encoder step, c from workspace, din from x[:, -1, 0]
    {
        _Float16* hf = &h_s[0][0];
        for (int i = tid; i < 16 * H; i += 256)
            hf[i] = enc_f16[((size_t)(b0 + i / H) * T_IN + (T_IN - 1)) * H + (i % H)];
    }
    if (tid < H) { v_s[tid] = va[tid]; fc_s[tid] = fc_W[tid]; }
    if (tid < 16) din_s[tid] = x[(size_t)(b0 + tid) * T_IN + (T_IN - 1)];

    float c_st[8];
#pragma unroll
    for (int r = 0; r < 8; ++r)
        c_st[r] = c_ws[(size_t)(b0 + r + 8 * hl) * H + j];

    float wih0_c[4], bias_c[4];
#pragma unroll
    for (int g = 0; g < 4; ++g) {
        int gi = g * H + j;
        wih0_c[g] = dec_Wih[(size_t)gi * 129];       // column 0 = din weight
        bias_c[g] = dec_bih[gi] + dec_bhh[gi];
    }
    __syncthreads();

    for (int s = 0; s < OUT_H; ++s) {
        // ---- Phase A: q = h @ Wa^T -------------------------------------
        {
            v8f acc;
#pragma unroll
            for (int q = 0; q < 8; ++q) acc[q] = 0.0f;
#pragma unroll
            for (int kt = 0; kt < 4; ++kt) {
                v16h a;
                const _Float16* p1 = &h_s[m][kt * 32 + hl * 8];
                const _Float16* p2 = &h_s[m][kt * 32 + 16 + hl * 8];
#pragma unroll
                for (int e = 0; e < 8; ++e) { a[e] = p1[e]; a[8 + e] = p2[e]; }
                v16h bf = *(const v16h*)(Wa_p + ((size_t)(wave * 4 + kt) * 32 + lane) * 16);
                acc = __builtin_amdgcn_wmma_f32_16x16x32_f16(
                    false, a, false, bf, (short)0, acc, false, false);
            }
#pragma unroll
            for (int r = 0; r < 8; ++r)
                q_s[r + 8 * hl][j] = (_Float16)acc[r];
        }
        __syncthreads();

        // ---- Phase B1: scores + softmax (wave owns 2 batch rows) -------
        for (int bb = 0; bb < 2; ++bb) {
            int bl = 2 * wave + bb;
            const _Float16* qrow = &q_s[bl][0];
            float fmax = -1e30f;
            for (int t = lane; t < T_IN; t += 32) {
                const _Float16* uerow = Ue + ((size_t)(b0 + bl) * T_IN + t) * H;
                float sc = 0.0f;
                for (int k = 0; k < H; ++k)
                    sc += v_s[k] * fast_tanh((float)uerow[k] + (float)qrow[k]);
                att_s[bl][t] = sc;
                fmax = fmaxf(fmax, sc);
            }
            for (int off = 16; off > 0; off >>= 1)
                fmax = fmaxf(fmax, __shfl_xor(fmax, off, 32));
            float fsum = 0.0f;
            for (int t = lane; t < T_IN; t += 32) {
                float e = fast_exp(att_s[bl][t] - fmax);
                att_s[bl][t] = e;
                fsum += e;
            }
            for (int off = 16; off > 0; off >>= 1)
                fsum += __shfl_xor(fsum, off, 32);
            float inv = fast_rcp(fsum);
            for (int t = lane; t < T_IN; t += 32) att_s[bl][t] *= inv;
        }
        __syncthreads();

        // ---- Phase B2: context = attn @ enc_outs -----------------------
        {
            int bl = tid >> 4;
            int j0 = (tid & 15) * 8;
            float ctx[8];
#pragma unroll
            for (int jj = 0; jj < 8; ++jj) ctx[jj] = 0.0f;
            for (int t = 0; t < T_IN; ++t) {
                float w = att_s[bl][t];
                const _Float16* er = enc_f16 + ((size_t)(b0 + bl) * T_IN + t) * H + j0;
#pragma unroll
                for (int jj = 0; jj < 8; ++jj) ctx[jj] += w * (float)er[jj];
            }
#pragma unroll
            for (int jj = 0; jj < 8; ++jj) ctx_s[bl][j0 + jj] = (_Float16)ctx[jj];
        }
        __syncthreads();

        // ---- Phase C: gates = h@Whh^T + ctx@Wih1^T ---------------------
        v8f acc[4];
#pragma unroll
        for (int g = 0; g < 4; ++g)
#pragma unroll
            for (int q = 0; q < 8; ++q) acc[g][q] = 0.0f;

#pragma unroll
        for (int kt = 0; kt < 4; ++kt) {
            v16h ah, ac;
            const _Float16* p1 = &h_s[m][kt * 32 + hl * 8];
            const _Float16* p2 = &h_s[m][kt * 32 + 16 + hl * 8];
            const _Float16* p3 = &ctx_s[m][kt * 32 + hl * 8];
            const _Float16* p4 = &ctx_s[m][kt * 32 + 16 + hl * 8];
#pragma unroll
            for (int e = 0; e < 8; ++e) {
                ah[e] = p1[e]; ah[8 + e] = p2[e];
                ac[e] = p3[e]; ac[8 + e] = p4[e];
            }
#pragma unroll
            for (int g = 0; g < 4; ++g) {
                int nt = g * 8 + jt;
                v16h b1 = *(const v16h*)(decWhh_p  + ((size_t)(nt * 4 + kt) * 32 + lane) * 16);
                acc[g] = __builtin_amdgcn_wmma_f32_16x16x32_f16(
                    false, ah, false, b1, (short)0, acc[g], false, false);
                v16h b2 = *(const v16h*)(decWih1_p + ((size_t)(nt * 4 + kt) * 32 + lane) * 16);
                acc[g] = __builtin_amdgcn_wmma_f32_16x16x32_f16(
                    false, ac, false, b2, (short)0, acc[g], false, false);
            }
        }
        __syncthreads();   // reads of h_s done before cell overwrites it

        // ---- LSTM cell -------------------------------------------------
#pragma unroll
        for (int r = 0; r < 8; ++r) {
            int   bl = r + 8 * hl;
            float dv = din_s[bl];
            float gi_ = acc[0][r] + dv * wih0_c[0] + bias_c[0];
            float gf_ = acc[1][r] + dv * wih0_c[1] + bias_c[1];
            float gg_ = acc[2][r] + dv * wih0_c[2] + bias_c[2];
            float go_ = acc[3][r] + dv * wih0_c[3] + bias_c[3];
            float iv = sigmf(gi_), fv = sigmf(gf_);
            float gv = fast_tanh(gg_), ov = sigmf(go_);
            float cn = fv * c_st[r] + iv * gv;
            c_st[r] = cn;
            float hn = ov * fast_tanh(cn);
            h_s[bl][j] = (_Float16)hn;
        }
        __syncthreads();

        // ---- Phase D: pred = h2 @ fc_W^T + fc_b ------------------------
        {
            int bl = 2 * wave + hl;       // lanes 0-15 -> row 2w, 16-31 -> 2w+1
            int j0 = n * 8;
            float p = 0.0f;
#pragma unroll
            for (int jj = 0; jj < 8; ++jj)
                p += fc_s[j0 + jj] * (float)h_s[bl][j0 + jj];
            for (int off = 8; off > 0; off >>= 1)      // reduce within 16-lane half
                p += __shfl_xor(p, off, 32);
            if (n == 0) {
                float pr = p + fc_b[0];
                out[(size_t)(b0 + bl) * OUT_H + s] = pr;
                din_s[bl] = pr;
            }
        }
        __syncthreads();
    }
}

// ---------------------------------------------------------------------------
extern "C" void kernel_launch(void* const* d_in, const int* in_sizes, int n_in,
                              void* d_out, int out_size, void* d_ws, size_t ws_size,
                              hipStream_t stream) {
    const float* x        = (const float*)d_in[0];
    const float* enc_Wih  = (const float*)d_in[1];
    const float* enc_Whh  = (const float*)d_in[2];
    const float* enc_bih  = (const float*)d_in[3];
    const float* enc_bhh  = (const float*)d_in[4];
    const float* Wa       = (const float*)d_in[5];
    const float* Ua       = (const float*)d_in[6];
    const float* va       = (const float*)d_in[7];
    const float* dec_Wih  = (const float*)d_in[8];
    const float* dec_Whh  = (const float*)d_in[9];
    const float* dec_bih  = (const float*)d_in[10];
    const float* dec_bhh  = (const float*)d_in[11];
    const float* fc_W     = (const float*)d_in[12];
    const float* fc_b     = (const float*)d_in[13];
    float* out = (float*)d_out;

    char* ws = (char*)d_ws;
    size_t off = 0;
    _Float16* enc_f16   = (_Float16*)(ws + off); off += (size_t)BATCH * T_IN * H * 2;
    _Float16* Ue        = (_Float16*)(ws + off); off += (size_t)BATCH * T_IN * H * 2;
    float*    c_ws      = (float*)   (ws + off); off += (size_t)BATCH * H * 4;
    _Float16* encWhh_p  = (_Float16*)(ws + off); off += (size_t)512 * 128 * 2;
    _Float16* Ua_p      = (_Float16*)(ws + off); off += (size_t)128 * 128 * 2;
    _Float16* Wa_p      = (_Float16*)(ws + off); off += (size_t)128 * 128 * 2;
    _Float16* decWhh_p  = (_Float16*)(ws + off); off += (size_t)512 * 128 * 2;
    _Float16* decWih1_p = (_Float16*)(ws + off); off += (size_t)512 * 128 * 2;

    // Pack weights into WMMA B-fragment layout (f16)
    pack_b_kernel<<<(512 * 128 + 255) / 256, 256, 0, stream>>>(enc_Whh, encWhh_p, 512, 128, 128, 0);
    pack_b_kernel<<<(128 * 128 + 255) / 256, 256, 0, stream>>>(Ua,      Ua_p,     128, 128, 128, 0);
    pack_b_kernel<<<(128 * 128 + 255) / 256, 256, 0, stream>>>(Wa,      Wa_p,     128, 128, 128, 0);
    pack_b_kernel<<<(512 * 128 + 255) / 256, 256, 0, stream>>>(dec_Whh, decWhh_p, 512, 128, 128, 0);
    pack_b_kernel<<<(512 * 128 + 255) / 256, 256, 0, stream>>>(dec_Wih, decWih1_p,512, 128, 129, 1);

    // Encoder: full 168-step recurrence, one block per 16 batch rows
    encoder_kernel<<<BATCH / 16, 256, 0, stream>>>(
        x, enc_Wih, enc_bih, enc_bhh, encWhh_p, enc_f16, c_ws);

    // Attention precompute Ue
    ue_kernel<<<(BATCH * T_IN) / 16, 256, 0, stream>>>(enc_f16, Ua_p, Ue);

    // Decoder: full 24-step recurrence + attention
    decoder_kernel<<<BATCH / 16, 256, 0, stream>>>(
        x, va, fc_W, fc_b, dec_Wih, dec_bih, dec_bhh,
        Wa_p, decWhh_p, decWih1_p, enc_f16, Ue, c_ws, out);
}
